// GramDiTBlock_11012296147496
// MI455X (gfx1250) — matrix operand order repaired and checked by reading, not compiled, not run on hardware
//
#include <hip/hip_runtime.h>
#include <math.h>

typedef unsigned short u16;
typedef __bf16 v16bf __attribute__((ext_vector_type(16)));
typedef float  v8f   __attribute__((ext_vector_type(8)));
typedef unsigned int u32x4 __attribute__((ext_vector_type(4)));
typedef int i32x8 __attribute__((ext_vector_type(8)));
typedef int i32x4 __attribute__((ext_vector_type(4)));

#define DEV __device__ __forceinline__

DEV u16 f2bf(float f) {
    unsigned int u = __float_as_uint(f);
    u += 0x7FFFu + ((u >> 16) & 1u);          // round-to-nearest-even
    return (u16)(u >> 16);
}

DEV float gelu_tanh(float x) {
    float x3 = x * x * x;
    return 0.5f * x * (1.f + tanhf(0.7978845608f * (x + 0.044715f * x3)));
}

DEV v8f zero8() {
    v8f z;
    #pragma unroll
    for (int i = 0; i < 8; ++i) z[i] = 0.f;
    return z;
}

// 32-bit LDS byte offset of a __shared__ object (for TDM D# lds_addr)
DEV unsigned lds_offset(const void* p) {
    return (unsigned)(unsigned long long)(__attribute__((address_space(3))) const void*)p;
}

// ---- WMMA fragment loads from LDS (bf16, contiguous k = 16*half + s for both A and B) ----
// A-matrix 16x32: lane<16 holds K0..15, lane>=16 holds K16..31
DEV v16bf ld_frag_a(const u16* rowp, int half) {
    union { uint4 u[2]; v16bf v; } un;
    const uint4* p = (const uint4*)rowp;      // row = 32 bf16 = 4 x 16B chunks
    un.u[0] = p[2 * half];
    un.u[1] = p[2 * half + 1];
    return un.v;
}
// B-matrix 32x16 staged as Bt[N][K]: lane<16 -> K0..15, lane>=16 -> K16..31
DEV v16bf ld_frag_b(const u16* rowp, int half) {
    union { uint4 u[2]; v16bf v; } un;
    const uint4* p = (const uint4*)rowp;
    un.u[0] = p[2 * half];
    un.u[1] = p[2 * half + 1];
    return un.v;
}

DEV v8f wmma_bf16(v16bf a, v16bf b, v8f c) {
    return __builtin_amdgcn_wmma_f32_16x16x32_bf16(false, a, false, b,
                                                   (short)0, c, false, false);
}

// ---- 256-thread block sum (wave32) ----
DEV float blk_sum256(float v, float* sb) {
    #pragma unroll
    for (int o = 16; o; o >>= 1) v += __shfl_xor(v, o, 32);
    int t = threadIdx.x;
    if ((t & 31) == 0) sb[t >> 5] = v;
    __syncthreads();
    if (t < 32) {
        float w = (t < 8) ? sb[t] : 0.f;
        #pragma unroll
        for (int o = 4; o; o >>= 1) w += __shfl_xor(w, o, 32);
        if (t == 0) sb[0] = w;
    }
    __syncthreads();
    float r = sb[0];
    __syncthreads();
    return r;
}

// ================= weight convert + transpose: f32 W[K][N] -> bf16 Wt[N][K]
__global__ __launch_bounds__(256) void k_convT(const float* __restrict__ W,
                                               u16* __restrict__ Wt,
                                               int K, int N) {
    int idx = blockIdx.x * 256 + threadIdx.x;
    if (idx >= K * N) return;
    int k = idx / N, n = idx % N;
    Wt[(size_t)n * K + k] = f2bf(W[idx]);
}

// ================= c_proj = silu(c) @ Wc + bc   (B x 6144)
__global__ __launch_bounds__(256) void k_cond(const float* __restrict__ c,
                                              const float* __restrict__ Wc,
                                              const float* __restrict__ bc,
                                              float* __restrict__ cproj) {
    __shared__ float sc[1024];
    int bb = blockIdx.x / 24;
    int j  = (blockIdx.x % 24) * 256 + threadIdx.x;
    for (int i = threadIdx.x; i < 1024; i += 256) {
        float v = c[bb * 1024 + i];
        sc[i] = v / (1.f + __expf(-v));
    }
    __syncthreads();
    float acc = bc[j];
    for (int d = 0; d < 1024; ++d) acc = fmaf(sc[d], Wc[(size_t)d * 6144 + j], acc);
    cproj[bb * 6144 + j] = acc;
}

// ================= LayerNorm + modulate -> bf16
__global__ __launch_bounds__(256) void k_ln_mod(const float* __restrict__ x,
                                                const float* __restrict__ cproj,
                                                u16* __restrict__ out) {
    __shared__ float red[16];
    int row = blockIdx.x, b = row >> 10, t = threadIdx.x;
    const float* xr = x + (size_t)row * 1024;
    float v[4], s = 0.f, sq = 0.f;
    #pragma unroll
    for (int i = 0; i < 4; ++i) { v[i] = xr[t + i * 256]; s += v[i]; sq += v[i] * v[i]; }
    float mean = blk_sum256(s, red) * (1.f / 1024.f);
    float msq  = blk_sum256(sq, red) * (1.f / 1024.f);
    float inv  = rsqrtf(msq - mean * mean + 1e-6f);
    const float* sh = cproj + b * 6144;
    const float* sc = cproj + b * 6144 + 1024;
    #pragma unroll
    for (int i = 0; i < 4; ++i) {
        int d = t + i * 256;
        float xn = (v[i] - mean) * inv;
        out[(size_t)row * 1024 + d] = f2bf(xn * (1.f + sc[d]) + sh[d]);
    }
}

// ================= bf16 WMMA GEMM: C[M,N] = A[M,K] * Bt[N,K]^T (+epilogue)
// B tile (128 x 32 bf16, LDS rows padded to 40 u16) is fetched by the Tensor
// Data Mover: D# pad_enable inserts 4 DWORDs after every 16 DWORDs stored,
// reproducing the stride-40 layout (see CDNA5 ISA).
// EPI 0: +bias -> bf16 ; EPI 1: gelu(+bias) -> bf16 ; EPI 2: resid + gate*(+bias) -> f32
template <int EPI>
__global__ __launch_bounds__(256) void k_gemm(const u16* __restrict__ A,
                                              const u16* __restrict__ Bt,
                                              const float* __restrict__ bias,
                                              const float* __restrict__ resid,
                                              const float* __restrict__ cproj,
                                              int gateOff,
                                              u16* __restrict__ outBF,
                                              float* __restrict__ outF,
                                              int K, int Ntot) {
    __shared__ u16 As[128 * 40];   // 128 rows x 32 bf16, stride 40
    __shared__ u16 Bs[128 * 40];
    const int t = threadIdx.x, wave = t >> 5, lane = t & 31;
    const int lm = lane & 15, half = lane >> 4;
    const int wm = wave >> 2, wn = wave & 3;         // 2 x 4 wave grid
    const int row0 = blockIdx.y * 128, col0 = blockIdx.x * 128;

    v8f acc[4][2];
    #pragma unroll
    for (int mi = 0; mi < 4; ++mi)
        #pragma unroll
        for (int ni = 0; ni < 2; ++ni) acc[mi][ni] = zero8();

    const int r = t >> 1, hh = t & 1;
    const size_t arow = (size_t)(row0 + r) * K;
    const unsigned bs_lds = lds_offset(&Bs[0]);

    for (int k0 = 0; k0 < K; k0 += 32) {
        __syncthreads();
        if (wave == 0) {
            // ---- TDM: load 128x32 bf16 weight tile -> LDS (padded rows) ----
            unsigned long long ga =
                (unsigned long long)(const void*)(Bt + (size_t)col0 * K + k0);
            u32x4 g0;
            g0[0] = 1u;                                       // count=1 (valid D#)
            g0[1] = bs_lds;                                   // lds_addr
            g0[2] = (unsigned)(ga & 0xFFFFFFFFu);             // global_addr lo
            g0[3] = (unsigned)((ga >> 32) & 0x1FFFFFFu) | (2u << 30); // hi | type=2
            i32x8 g1;
            g1[0] = (1 << 16) | (1 << 20) | (3 << 22) | (3 << 25);
            //        data=2B    pad_en     intvl=16dw  amount=4dw
            g1[1] = (K & 0xFFFF) << 16;                       // tensor_dim0 lo
            g1[2] = (Ntot & 0xFFFF) << 16;                    // dim0 hi=0 | tensor_dim1 lo
            g1[3] = (32 << 16);                               // dim1 hi=0 | tile_dim0=32
            g1[4] = 128;                                      // tile_dim1=128, tile_dim2=0
            g1[5] = K;                                        // tensor_dim0_stride lo
            g1[6] = 0;
            g1[7] = 0;
            i32x4 gz4 = {0, 0, 0, 0};
            i32x8 gz8 = {0, 0, 0, 0, 0, 0, 0, 0};
            __builtin_amdgcn_tensor_load_to_lds(g0, g1, gz4, gz4, gz8, 0);
        }
        {   // A tile via vector copy (overlaps the TDM transfer)
            const uint4* sa = (const uint4*)(A + arow + k0 + hh * 16);
            uint4* da = (uint4*)(&As[r * 40 + hh * 16]);
            da[0] = sa[0]; da[1] = sa[1];
            if (k0 + 32 < K)
                __builtin_prefetch(A + arow + k0 + 32 + hh * 16, 0, 0);
        }
        if (wave == 0) __builtin_amdgcn_s_wait_tensorcnt(0);
        __syncthreads();

        v16bf afr[4], bfr[2];
        #pragma unroll
        for (int mi = 0; mi < 4; ++mi)
            afr[mi] = ld_frag_a(&As[(wm * 64 + mi * 16 + lm) * 40], half);
        #pragma unroll
        for (int ni = 0; ni < 2; ++ni)
            bfr[ni] = ld_frag_b(&Bs[(wn * 32 + ni * 16 + lm) * 40], half);
        #pragma unroll
        for (int mi = 0; mi < 4; ++mi)
            #pragma unroll
            for (int ni = 0; ni < 2; ++ni)
                acc[mi][ni] = wmma_bf16(afr[mi], bfr[ni], acc[mi][ni]);
    }

    #pragma unroll
    for (int mi = 0; mi < 4; ++mi)
        #pragma unroll
        for (int ni = 0; ni < 2; ++ni)
            #pragma unroll
            for (int v = 0; v < 8; ++v) {
                int grow = row0 + wm * 64 + mi * 16 + v + 8 * half;
                int gcol = col0 + wn * 32 + ni * 16 + lm;
                float val = acc[mi][ni][v] + bias[gcol];
                size_t o = (size_t)grow * Ntot + gcol;
                if (EPI == 0) {
                    outBF[o] = f2bf(val);
                } else if (EPI == 1) {
                    outBF[o] = f2bf(gelu_tanh(val));
                } else {
                    float g = cproj[(grow >> 10) * 6144 + gateOff + gcol];
                    outF[o] = resid[o] + g * val;
                }
            }
}

// ================= flash attention (per b,h; 128 query rows / block)
__global__ __launch_bounds__(256) void k_attn(const u16* __restrict__ qkv,
                                              u16* __restrict__ attn_out) {
    const int bh = blockIdx.y, b = bh >> 4, h = bh & 15;
    const int n0 = blockIdx.x * 128;
    const int t = threadIdx.x, wave = t >> 5, lane = t & 31;
    const int lm = lane & 15, half = lane >> 4;

    __shared__ u16 Qs[128 * 72];     // 128 x 64, stride 72
    __shared__ u16 Ks[32 * 72];      // 32 keys x 64
    __shared__ u16 Vt[64 * 40];      // 64 hd x 32 keys, stride 40
    __shared__ u16 Ps[8 * 16 * 40];  // per-wave 16 x 32 P tile

    const size_t rowbase = (size_t)b * 1024 * 3072;

    {   // stage Q tile
        int r = t >> 1, hh = t & 1;
        const uint4* src = (const uint4*)(qkv + rowbase + (size_t)(n0 + r) * 3072 + h * 64 + hh * 32);
        uint4* dst = (uint4*)(&Qs[r * 72 + hh * 32]);
        dst[0] = src[0]; dst[1] = src[1]; dst[2] = src[2]; dst[3] = src[3];
    }
    __syncthreads();

    v16bf aq[2];
    {
        int qrow = wave * 16 + lm;
        aq[0] = ld_frag_a(&Qs[qrow * 72 + 0], half);
        aq[1] = ld_frag_a(&Qs[qrow * 72 + 32], half);
    }

    v8f oacc[4];
    #pragma unroll
    for (int nc = 0; nc < 4; ++nc) oacc[nc] = zero8();
    float mrow[8], lrow[8];
    #pragma unroll
    for (int v = 0; v < 8; ++v) { mrow[v] = -1e30f; lrow[v] = 0.f; }

    u16* Pw = &Ps[wave * 640];

    for (int kt = 0; kt < 1024; kt += 32) {
        __syncthreads();
        {   // stage K (row-major) and V (transposed)
            int r = t >> 3, seg = t & 7;
            const uint4* sk = (const uint4*)(qkv + rowbase + (size_t)(kt + r) * 3072 + 1024 + h * 64 + seg * 8);
            *(uint4*)(&Ks[r * 72 + seg * 8]) = sk[0];
            int kk = t & 31, dseg = t >> 5;
            const uint4* sv = (const uint4*)(qkv + rowbase + (size_t)(kt + kk) * 3072 + 2048 + h * 64 + dseg * 8);
            uint4 vv = sv[0];
            const u16* ve = (const u16*)&vv;
            #pragma unroll
            for (int i = 0; i < 8; ++i) Vt[(dseg * 8 + i) * 40 + kk] = ve[i];
        }
        __syncthreads();

        // S = Q K^T  (two 16x16 tiles over 32 keys, k = 64 in two steps)
        v8f sacc[2] = { zero8(), zero8() };
        #pragma unroll
        for (int s = 0; s < 2; ++s) {
            #pragma unroll
            for (int ni = 0; ni < 2; ++ni) {
                v16bf bk = ld_frag_b(&Ks[(ni * 16 + lm) * 72 + s * 32], half);
                sacc[ni] = wmma_bf16(aq[s], bk, sacc[ni]);
            }
        }

        // online softmax (row = v + 8*half; reduce across 16-lane half)
        #pragma unroll
        for (int v = 0; v < 8; ++v) {
            float s0 = sacc[0][v] * 0.125f, s1 = sacc[1][v] * 0.125f;
            float mx = fmaxf(s0, s1);
            #pragma unroll
            for (int o = 8; o; o >>= 1) mx = fmaxf(mx, __shfl_xor(mx, o, 32));
            float mnew = fmaxf(mrow[v], mx);
            float p0 = __expf(s0 - mnew), p1 = __expf(s1 - mnew);
            float rs = p0 + p1;
            #pragma unroll
            for (int o = 8; o; o >>= 1) rs += __shfl_xor(rs, o, 32);
            float alpha = __expf(mrow[v] - mnew);
            mrow[v] = mnew;
            lrow[v] = lrow[v] * alpha + rs;
            #pragma unroll
            for (int nc = 0; nc < 4; ++nc) oacc[nc][v] *= alpha;
            int prow = v + 8 * half;
            Pw[prow * 40 + lm]      = f2bf(p0);
            Pw[prow * 40 + 16 + lm] = f2bf(p1);
        }
        __syncthreads();

        // O += P V
        v16bf ap = ld_frag_a(&Pw[lm * 40], half);
        #pragma unroll
        for (int nc = 0; nc < 4; ++nc) {
            v16bf bv = ld_frag_b(&Vt[(nc * 16 + lm) * 40], half);
            oacc[nc] = wmma_bf16(ap, bv, oacc[nc]);
        }
    }

    #pragma unroll
    for (int v = 0; v < 8; ++v) {
        float inv = 1.f / lrow[v];
        int grow = b * 1024 + n0 + wave * 16 + v + 8 * half;
        #pragma unroll
        for (int nc = 0; nc < 4; ++nc)
            attn_out[(size_t)grow * 1024 + h * 64 + nc * 16 + lm] = f2bf(oacc[nc][v] * inv);
    }
}

// ================= T[row][r] = sum_d X[row][d] * A[d][r]   (rank 32)
__global__ __launch_bounds__(256) void k_xA(const float* __restrict__ X,
                                            const float* __restrict__ A,
                                            float* __restrict__ T) {
    int idx = blockIdx.x * 256 + threadIdx.x;
    int row = idx >> 5, rr = idx & 31;
    const float* xr = X + (size_t)row * 1024;
    float acc = 0.f;
    for (int d = 0; d < 1024; ++d) acc = fmaf(xr[d], A[d * 32 + rr], acc);
    T[row * 32 + rr] = acc;
}

__global__ __launch_bounds__(256) void k_xA_cls(const float* __restrict__ X,
                                                const float* __restrict__ cls,
                                                const float* __restrict__ C,
                                                float* __restrict__ T) {
    int idx = blockIdx.x * 256 + threadIdx.x;
    int row = idx >> 5, rr = idx & 31, b = row >> 10;
    const float* xr = X + (size_t)row * 1024;
    const float* cl = cls + b * 1024;
    float acc = 0.f;
    for (int d = 0; d < 1024; ++d) acc = fmaf(xr[d] * cl[d], C[d * 32 + rr], acc);
    T[row * 32 + rr] = acc;
}

// ================= X' = rms_s(T @ B_s) + Y    (Y updated in place)
__global__ __launch_bounds__(256) void k_gram_s(const float* __restrict__ T,
                                                const float* __restrict__ Bsw,
                                                const float* __restrict__ scale,
                                                float* __restrict__ Y) {
    __shared__ float sT[32];
    __shared__ float red[16];
    int row = blockIdx.x, t = threadIdx.x;
    if (t < 32) sT[t] = T[row * 32 + t];
    __syncthreads();
    float g[4], sq = 0.f;
    #pragma unroll
    for (int i = 0; i < 4; ++i) {
        int d = t + i * 256;
        float a = 0.f;
        #pragma unroll
        for (int rr = 0; rr < 32; ++rr) a = fmaf(sT[rr], Bsw[rr * 1024 + d], a);
        g[i] = a; sq += a * a;
    }
    float inv = rsqrtf(blk_sum256(sq, red) * (1.f / 1024.f) + 1e-6f);
    #pragma unroll
    for (int i = 0; i < 4; ++i) {
        int d = t + i * 256;
        size_t o = (size_t)row * 1024 + d;
        Y[o] = scale[d] * g[i] * inv + Y[o];
    }
}

// ================= Z = rms_c(T2 @ D_r); z_norm = modulate(LN(Z)) -> bf16
__global__ __launch_bounds__(256) void k_gram_c(const float* __restrict__ T2,
                                                const float* __restrict__ Dr,
                                                const float* __restrict__ scalec,
                                                const float* __restrict__ cproj,
                                                float* __restrict__ Z,
                                                u16* __restrict__ znorm) {
    __shared__ float sT[32];
    __shared__ float red[16];
    int row = blockIdx.x, b = row >> 10, t = threadIdx.x;
    if (t < 32) sT[t] = T2[row * 32 + t];
    __syncthreads();
    float g[4], sq = 0.f;
    #pragma unroll
    for (int i = 0; i < 4; ++i) {
        int d = t + i * 256;
        float a = 0.f;
        #pragma unroll
        for (int rr = 0; rr < 32; ++rr) a = fmaf(sT[rr], Dr[rr * 1024 + d], a);
        g[i] = a; sq += a * a;
    }
    float inv = rsqrtf(blk_sum256(sq, red) * (1.f / 1024.f) + 1e-6f);
    float z[4], s = 0.f, s2 = 0.f;
    #pragma unroll
    for (int i = 0; i < 4; ++i) {
        int d = t + i * 256;
        z[i] = scalec[d] * g[i] * inv;
        s += z[i]; s2 += z[i] * z[i];
        Z[(size_t)row * 1024 + d] = z[i];
    }
    float mean = blk_sum256(s, red) * (1.f / 1024.f);
    float msq  = blk_sum256(s2, red) * (1.f / 1024.f);
    float linv = rsqrtf(msq - mean * mean + 1e-6f);
    const float* shm = cproj + b * 6144 + 3072;
    const float* scm = cproj + b * 6144 + 4096;
    #pragma unroll
    for (int i = 0; i < 4; ++i) {
        int d = t + i * 256;
        float zn = (z[i] - mean) * linv;
        znorm[(size_t)row * 1024 + d] = f2bf(zn * (1.f + scm[d]) + shm[d]);
    }
}

// =====================================================================
extern "C" void kernel_launch(void* const* d_in, const int* in_sizes, int n_in,
                              void* d_out, int out_size, void* d_ws, size_t ws_size,
                              hipStream_t stream) {
    (void)in_sizes; (void)n_in; (void)out_size; (void)ws_size;
    const float* x    = (const float*)d_in[0];
    const float* c    = (const float*)d_in[1];
    const float* cls  = (const float*)d_in[2];
    const float* Wc   = (const float*)d_in[3];
    const float* bc   = (const float*)d_in[4];
    const float* Wqkv = (const float*)d_in[5];
    const float* bqkv = (const float*)d_in[6];
    const float* Wo   = (const float*)d_in[7];
    const float* bo   = (const float*)d_in[8];
    const float* A_s  = (const float*)d_in[9];
    const float* B_s  = (const float*)d_in[10];
    const float* rmss = (const float*)d_in[11];
    const float* C_r  = (const float*)d_in[12];
    const float* D_r  = (const float*)d_in[13];
    const float* rmsc = (const float*)d_in[14];
    const float* W1   = (const float*)d_in[15];
    const float* b1   = (const float*)d_in[16];
    const float* W2   = (const float*)d_in[17];
    const float* b2   = (const float*)d_in[18];

    char* ws = (char*)d_ws;
    u16*   wqkvT = (u16*)  (ws + 0);          // 1024x3072 bf16 (transposed)
    u16*   woT   = (u16*)  (ws + 6291456);    // 1024x1024
    u16*   w1T   = (u16*)  (ws + 8388608);    // 1024x4096
    u16*   w2T   = (u16*)  (ws + 16777216);   // 4096x1024
    float* cproj = (float*)(ws + 25165824);   // 4x6144
    u16*   xnorm = (u16*)  (ws + 25264128);   // 4096x1024 bf16 (also attn_out)
    u16*   qkv   = (u16*)  (ws + 33652736);   // 4096x3072 bf16
    float* Ybuf  = (float*)(ws + 58818560);   // 4096x1024 f32 (Y then X')
    float* Tbuf  = (float*)(ws + 75595776);   // 4096x32
    float* T2buf = (float*)(ws + 76120064);   // 4096x32
    float* Zbuf  = (float*)(ws + 76644352);   // 4096x1024 f32
    u16*   znorm = (u16*)  (ws + 93421568);   // 4096x1024 bf16
    u16*   hbuf  = xnorm;                     // 4096x4096 bf16 overlays xnorm+qkv

    // weights -> bf16 transposed
    k_convT<<<(1024 * 3072 + 255) / 256, 256, 0, stream>>>(Wqkv, wqkvT, 1024, 3072);
    k_convT<<<(1024 * 1024 + 255) / 256, 256, 0, stream>>>(Wo,   woT,   1024, 1024);
    k_convT<<<(1024 * 4096 + 255) / 256, 256, 0, stream>>>(W1,   w1T,   1024, 4096);
    k_convT<<<(4096 * 1024 + 255) / 256, 256, 0, stream>>>(W2,   w2T,   4096, 1024);

    // conditioning + LN/modulate
    k_cond<<<96, 256, 0, stream>>>(c, Wc, bc, cproj);
    k_ln_mod<<<4096, 256, 0, stream>>>(x, cproj, xnorm);

    // QKV GEMM (bias -> bf16)
    k_gemm<0><<<dim3(24, 32), 256, 0, stream>>>(xnorm, wqkvT, bqkv,
                                                nullptr, nullptr, 0,
                                                qkv, nullptr, 1024, 3072);
    // attention
    k_attn<<<dim3(8, 64), 256, 0, stream>>>(qkv, xnorm /*attn_out*/);

    // out-proj, Y = x + gate_a * (attn @ Wo + bo)
    k_gemm<2><<<dim3(8, 32), 256, 0, stream>>>(xnorm, woT, bo,
                                               x, cproj, 2048,
                                               nullptr, Ybuf, 1024, 1024);
    // gram-s branch: X' = rms((x@A_s)@B_s) + Y
    k_xA<<<512, 256, 0, stream>>>(x, A_s, Tbuf);
    k_gram_s<<<4096, 256, 0, stream>>>(Tbuf, B_s, rmss, Ybuf);

    // cross-gram: Z = rms((X' @ (cls*C_r)) @ D_r); z_norm = modulate(LN(Z))
    k_xA_cls<<<512, 256, 0, stream>>>(Ybuf, cls, C_r, T2buf);
    k_gram_c<<<4096, 256, 0, stream>>>(T2buf, D_r, rmsc, cproj, Zbuf, znorm);

    // MLP
    k_gemm<1><<<dim3(32, 32), 256, 0, stream>>>(znorm, w1T, b1,
                                                nullptr, nullptr, 0,
                                                hbuf, nullptr, 1024, 4096);
    k_gemm<2><<<dim3(8, 32), 256, 0, stream>>>(hbuf, w2T, b2,
                                               Zbuf, cproj, 5120,
                                               nullptr, (float*)d_out, 4096, 1024);
}
